// EncodeProcessDecode_48661979464040
// MI455X (gfx1250) — matrix operand order repaired and checked by reading, not compiled, not run on hardware
//
#include <hip/hip_runtime.h>
#include <string.h>

#define GN_NODES  16000
#define GN_EDGES  64000
#define GN_GRAPHS 8

typedef __attribute__((ext_vector_type(16))) __bf16 bf16x16;
typedef __attribute__((ext_vector_type(8)))  float  floatx8;

// ---------------------------------------------------------------------------
// A-matrix "gathered concat" descriptor: row r of A is the concatenation of up
// to 8 segments; each segment reads ptr[ ind? ind[r] : r ][0..width).
// ---------------------------------------------------------------------------
struct SegDesc {
  const float* ptr[8];
  const int*   ind[8];
  int start[9];   // cumulative column offsets; start[nseg] == din
  int width[8];   // row stride of each segment source
  int nseg;
};

__device__ __forceinline__ unsigned short f2bf_bits(float f) {
  unsigned int u = __builtin_bit_cast(unsigned int, f);
  u = (u + 0x7FFFu + ((u >> 16) & 1u)) >> 16;   // RNE
  return (unsigned short)u;
}

// B-fragment loads + NT independent WMMAs for one 32-wide k-window.
// wcol = Wt + l15*Kpad + hi*8 ; pass wcol + k0.
template <int NT>
__device__ __forceinline__ void gn_bdot(const bf16x16& a,
                                        const unsigned short* __restrict__ wk,
                                        size_t Kpad, floatx8* acc)
{
#pragma unroll
  for (int t = 0; t < NT; ++t) {
    union { uint4 u[2]; bf16x16 v; } bu;
    const uint4* bp = (const uint4*)(wk + (size_t)t * 16 * Kpad);
    bu.u[0] = bp[0];
    bu.u[1] = bp[2];
    acc[t] = __builtin_amdgcn_wmma_f32_16x16x32_bf16(false, a, false, bu.v,
                                                     (short)0, acc[t], false, false);
  }
}

// ---------------------------------------------------------------------------
// ALIGNED kernel: every segment start is a multiple of 32, so segment bounds
// are 32-aligned and each k-window lies in ONE segment.  Loop nest is
// segment-major: descriptor fields, gathered row index and row pointer are
// inner-loop invariant (held in registers, no per-window kernarg traffic).
// One wave owns a 16-row strip and all NT column tiles.
// A frag (16-bit A 16x32): lanes 0-15 row M=l, K = k0+{0..7,16..23};
//                          lanes 16-31 row M=l, K = k0+{8..15,24..31}.
// B frag: lane n<16 col N=n K=k0+{0..7,16..23}; lane n+16 col N=n K=k0+{8..15,24..31}.
// ---------------------------------------------------------------------------
template <int NT>
__global__ __launch_bounds__(256)
void gn_gemm_wmma_al(SegDesc sd, const unsigned short* __restrict__ Wt,
                     const float* __restrict__ bias, float* __restrict__ out,
                     int M, int N, int Kpad, int ldo, int mtot, int relu)
{
  const int lane = threadIdx.x & 31;
  const int wv   = threadIdx.x >> 5;
  const int tm   = blockIdx.x * (blockDim.x >> 5) + wv;
  if (tm >= mtot) return;                 // wave-uniform exit (EXEC stays full)
  const int hi  = lane >> 4;
  const int l15 = lane & 15;
  const int r   = tm * 16 + l15;
  const int rr  = (r < M) ? r : 0;        // rows >= M compute garbage, never stored
  const int c0  = hi ? 8 : 0;
  const unsigned short* __restrict__ wcol = Wt + (size_t)l15 * Kpad + (hi ? 8 : 0);

  floatx8 acc[NT] = {};

  for (int s = 0; s < sd.nseg; ++s) {
    const float* __restrict__ base = sd.ptr[s];
    const int  segst = sd.start[s];
    const int  segen = sd.start[s + 1];
    const int  w     = sd.width[s];
    const int* ip    = sd.ind[s];
    const int  ridx  = ip ? ip[rr] : rr;                 // one gather per segment
    const float* __restrict__ rowp = base + (size_t)ridx * w;
    const int kfull = segst + ((segen - segst) & ~31);   // 32-aligned span

    for (int k0 = segst; k0 < kfull; k0 += 32) {         // full windows: no guards
      const float* p = rowp + (k0 - segst) + c0;
      const float4* q0 = (const float4*)p;
      const float4* q1 = (const float4*)(p + 16);
      const float4 a0 = q0[0], a1 = q0[1];
      const float4 b0 = q1[0], b1 = q1[1];
      union { unsigned short h[16]; bf16x16 v; } au;
      au.h[0]  = f2bf_bits(a0.x); au.h[1]  = f2bf_bits(a0.y);
      au.h[2]  = f2bf_bits(a0.z); au.h[3]  = f2bf_bits(a0.w);
      au.h[4]  = f2bf_bits(a1.x); au.h[5]  = f2bf_bits(a1.y);
      au.h[6]  = f2bf_bits(a1.z); au.h[7]  = f2bf_bits(a1.w);
      au.h[8]  = f2bf_bits(b0.x); au.h[9]  = f2bf_bits(b0.y);
      au.h[10] = f2bf_bits(b0.z); au.h[11] = f2bf_bits(b0.w);
      au.h[12] = f2bf_bits(b1.x); au.h[13] = f2bf_bits(b1.y);
      au.h[14] = f2bf_bits(b1.z); au.h[15] = f2bf_bits(b1.w);
      gn_bdot<NT>(au.v, wcol + k0, Kpad, acc);
    }

    if (kfull < segen) {                 // at most one guarded tail window
      const float* p = rowp + (kfull - segst);
      union { unsigned short h[16]; bf16x16 v; } au;
#pragma unroll
      for (int j = 0; j < 8; ++j) {
        const int k1 = kfull + c0 + j;
        const int k2 = k1 + 16;
        au.h[j]     = (k1 < segen) ? f2bf_bits(p[c0 + j])      : (unsigned short)0;
        au.h[j + 8] = (k2 < segen) ? f2bf_bits(p[c0 + 16 + j]) : (unsigned short)0;
      }
      gn_bdot<NT>(au.v, wcol + kfull, Kpad, acc);
    }
  }

  const int rowbase = tm * 16 + (hi ? 8 : 0);
#pragma unroll
  for (int t = 0; t < NT; ++t) {
    const int ncol = t * 16 + l15;
    if (ncol < N) {
      const float bv = bias[ncol];
#pragma unroll
      for (int i = 0; i < 8; ++i) {
        const int row = rowbase + i;
        if (row < M) {
          float v = acc[t][i] + bv;
          if (relu) v = fmaxf(v, 0.0f);
          out[(size_t)row * ldo + ncol] = v;
        }
      }
    }
  }
}

// ---------------------------------------------------------------------------
// GENERIC kernel: per-lane segment resolution (irregular widths/starts).
// Used only for encoder edge/node, output-node L1 and output-global L1.
// ---------------------------------------------------------------------------
__device__ __forceinline__ float fetch_one(const SegDesc& sd, const int* rowidx, int k) {
  float f = 0.0f;
  bool got = false;
#pragma unroll
  for (int i = 0; i < 8; ++i) {
    if (i < sd.nseg && !got && k < sd.start[i + 1]) {
      f = sd.ptr[i][(size_t)rowidx[i] * sd.width[i] + (k - sd.start[i])];
      got = true;
    }
  }
  return got ? f : 0.0f;
}

template <int NT>
__global__ __launch_bounds__(256)
void gn_gemm_wmma_gn(SegDesc sd, const unsigned short* __restrict__ Wt,
                     const float* __restrict__ bias, float* __restrict__ out,
                     int M, int N, int Kpad, int ldo, int mtot, int relu)
{
  const int lane = threadIdx.x & 31;
  const int wv   = threadIdx.x >> 5;
  const int tm   = blockIdx.x * (blockDim.x >> 5) + wv;
  if (tm >= mtot) return;
  const int hi  = lane >> 4;
  const int l15 = lane & 15;
  const int r   = tm * 16 + l15;
  const bool rv = (r < M);
  const int  rr = rv ? r : 0;
  int rowidx[8];
#pragma unroll
  for (int s = 0; s < 8; ++s) {
    const int* ip = (s < sd.nseg) ? sd.ind[s] : (const int*)0;
    rowidx[s] = ip ? ip[rr] : rr;
  }

  floatx8 acc[NT] = {};

  for (int k0 = 0; k0 < Kpad; k0 += 32) {
    union { unsigned short h[16]; bf16x16 v; } au;
    const int cbase = k0 + (hi ? 8 : 0);
#pragma unroll
    for (int j = 0; j < 8; ++j) {
      au.h[j]     = rv ? f2bf_bits(fetch_one(sd, rowidx, cbase + j))      : (unsigned short)0;
      au.h[j + 8] = rv ? f2bf_bits(fetch_one(sd, rowidx, cbase + 16 + j)) : (unsigned short)0;
    }
    const unsigned short* wk = Wt + (size_t)l15 * Kpad + (hi ? 8 : 0) + k0;
    gn_bdot<NT>(au.v, wk, Kpad, acc);
  }

  const int rowbase = tm * 16 + (hi ? 8 : 0);
#pragma unroll
  for (int t = 0; t < NT; ++t) {
    const int ncol = t * 16 + l15;
    if (ncol < N) {
      const float bv = bias[ncol];
#pragma unroll
      for (int i = 0; i < 8; ++i) {
        const int row = rowbase + i;
        if (row < M) {
          float v = acc[t][i] + bv;
          if (relu) v = fmaxf(v, 0.0f);
          out[(size_t)row * ldo + ncol] = v;
        }
      }
    }
  }
}

// Row-wise LayerNorm over N=128, one wave per row.
__global__ __launch_bounds__(256)
void gn_layernorm128(float* __restrict__ t, const float* __restrict__ g,
                     const float* __restrict__ b, int M)
{
  const int lane = threadIdx.x & 31;
  const int wv   = threadIdx.x >> 5;
  const int row  = blockIdx.x * 8 + wv;
  if (row >= M) return;
  float x[4];
  float s = 0.0f;
#pragma unroll
  for (int i = 0; i < 4; ++i) { x[i] = t[(size_t)row * 128 + lane + 32 * i]; s += x[i]; }
#pragma unroll
  for (int o = 16; o > 0; o >>= 1) s += __shfl_xor(s, o, 32);
  const float mu = s * (1.0f / 128.0f);
  float v = 0.0f;
#pragma unroll
  for (int i = 0; i < 4; ++i) { float d = x[i] - mu; v += d * d; }
#pragma unroll
  for (int o = 16; o > 0; o >>= 1) v += __shfl_xor(v, o, 32);
  const float inv = rsqrtf(v * (1.0f / 128.0f) + 1e-5f);
#pragma unroll
  for (int i = 0; i < 4; ++i) {
    const int c = lane + 32 * i;
    t[(size_t)row * 128 + c] = (x[i] - mu) * inv * g[c] + b[c];
  }
}

// dst[idx[i]][c] += src[i][c]  (segment_sum via float atomics)
__global__ __launch_bounds__(256)
void gn_scatter_add(const float* __restrict__ src, const int* __restrict__ idx,
                    float* __restrict__ dst, int n, int w)
{
  const long long t = (long long)blockIdx.x * blockDim.x + threadIdx.x;
  if (t >= (long long)n * w) return;
  const int i = (int)(t / w);
  const int c = (int)(t - (long long)i * w);
  atomicAdd(&dst[(size_t)idx[i] * w + c], src[(size_t)i * w + c]);
}

__global__ __launch_bounds__(256)
void gn_zero_f32(float* __restrict__ p, long long n)
{
  const long long t = (long long)blockIdx.x * blockDim.x + threadIdx.x;
  if (t < n) p[t] = 0.0f;
}

// eb[i] = batch[row[i]]
__global__ __launch_bounds__(256)
void gn_edge_batch(const int* __restrict__ batch, const int* __restrict__ row,
                   int* __restrict__ eb, int n)
{
  const int i = blockIdx.x * blockDim.x + threadIdx.x;
  if (i < n) eb[i] = batch[row[i]];
}

// Wt[Npad][Kpad] (bf16) = transpose(W[K][N]) with zero padding.
__global__ __launch_bounds__(256)
void gn_wt_cast(const float* __restrict__ W, unsigned short* __restrict__ Wt,
                int K, int N, int Kpad, int Npad)
{
  const long long t = (long long)blockIdx.x * blockDim.x + threadIdx.x;
  if (t >= (long long)Kpad * Npad) return;
  const int nn = (int)(t / Kpad);
  const int kk = (int)(t - (long long)nn * Kpad);
  unsigned short v = 0;
  if (nn < N && kk < K) v = f2bf_bits(W[(size_t)kk * N + nn]);
  Wt[t] = v;
}

// ---------------------------------------------------------------------------
// Host-side orchestration
// ---------------------------------------------------------------------------
struct Seg { const float* p; int w; };

struct SegBuild {
  SegDesc d;
  SegBuild() { memset(&d, 0, sizeof(d)); }
  void add(const float* p, int w, const int* ind) {
    const int s = d.nseg;
    d.ptr[s] = p; d.ind[s] = ind; d.width[s] = w;
    d.start[s + 1] = d.start[s] + w;
    d.nseg = s + 1;
  }
};

struct MLPW {
  const float *W1, *W2, *b1, *b2, *ln_b, *ln_g;
  unsigned short *W1t, *W2t;
  int din, lat, dout;
  bool ln;
};

static inline int rnd32(int x) { return (x + 31) & ~31; }
static inline int rnd16(int x) { return (x + 15) & ~15; }

static void launch_gemm(const SegDesc& sd, int K, const unsigned short* Wt,
                        const float* bias, float* out, int M, int N, int ldo,
                        bool relu, hipStream_t st)
{
  const int Kpad = rnd32(K);
  const int mtot = (M + 15) >> 4;
  const int blocks = (mtot + 7) >> 3;
  const int nt = (N + 15) >> 4;
  const int rl = relu ? 1 : 0;
  bool aligned = true;
  for (int i = 0; i < sd.nseg; ++i)
    if (sd.start[i] & 31) aligned = false;
#define GN_AL(NTV) gn_gemm_wmma_al<NTV><<<blocks, 256, 0, st>>>(sd, Wt, bias, out, M, N, Kpad, ldo, mtot, rl)
#define GN_GN(NTV) gn_gemm_wmma_gn<NTV><<<blocks, 256, 0, st>>>(sd, Wt, bias, out, M, N, Kpad, ldo, mtot, rl)
  if (aligned) {
    switch (nt) {
      case 1:  GN_AL(1); break;
      case 5:  GN_AL(5); break;
      default: GN_AL(8); break;
    }
  } else {
    switch (nt) {
      case 5:  GN_GN(5); break;
      default: GN_GN(8); break;
    }
  }
#undef GN_AL
#undef GN_GN
}

static void launch_zero(float* p, long long n, hipStream_t st) {
  gn_zero_f32<<<(unsigned)((n + 255) / 256), 256, 0, st>>>(p, n);
}
static void launch_scatter(const float* src, const int* idx, float* dst, int n, int w,
                           hipStream_t st) {
  const long long tot = (long long)n * w;
  gn_scatter_add<<<(unsigned)((tot + 255) / 256), 256, 0, st>>>(src, idx, dst, n, w);
}

static void run_block(const MLPW& me, const MLPW& mn, const MLPW& mg,
                      const Seg* ns, int nns, const Seg* es, int nes,
                      const Seg* gs, int ngs,
                      const int* row, const int* col, const int* batch, const int* eb,
                      float* e_out, float* n_out, float* g_out,
                      float* He, float* Hn, float* Hg,
                      float* aggS, float* aggR, float* n2g, float* e2g,
                      bool act_final, hipStream_t st)
{
  // Hidden row strides padded to 16 for aligned float4 rows (lat=65 -> 80).
  const int He_ld = rnd16(me.lat);
  const int Hn_ld = rnd16(mn.lat);
  const int Hg_ld = rnd16(mg.lat);

  // ---- edge MLP: concat(recv=n[col], send=n[row], e, g[edge_batch]) ----
  SegBuild A;
  for (int i = 0; i < nns; ++i) A.add(ns[i].p, ns[i].w, col);
  for (int i = 0; i < nns; ++i) A.add(ns[i].p, ns[i].w, row);
  for (int i = 0; i < nes; ++i) A.add(es[i].p, es[i].w, (const int*)0);
  for (int i = 0; i < ngs; ++i) A.add(gs[i].p, gs[i].w, eb);
  launch_gemm(A.d, me.din, me.W1t, me.b1, He, GN_EDGES, me.lat, He_ld, true, st);
  // Layer 2: declare K = padded stride; extra A columns hit zero rows of Wt.
  SegBuild A2; A2.add(He, He_ld, (const int*)0);
  launch_gemm(A2.d, He_ld, me.W2t, me.b2, e_out, GN_EDGES, me.dout, me.dout,
              act_final, st);
  if (me.ln)
    gn_layernorm128<<<(GN_EDGES + 7) / 8, 256, 0, st>>>(e_out, me.ln_g, me.ln_b, GN_EDGES);

  // ---- edge -> node aggregation ----
  launch_zero(aggS, (long long)GN_NODES * me.dout, st);
  launch_zero(aggR, (long long)GN_NODES * me.dout, st);
  launch_scatter(e_out, row, aggS, GN_EDGES, me.dout, st);
  launch_scatter(e_out, col, aggR, GN_EDGES, me.dout, st);

  // ---- node MLP: concat(n, g[batch], recv_agg, send_agg) ----
  SegBuild B;
  for (int i = 0; i < nns; ++i) B.add(ns[i].p, ns[i].w, (const int*)0);
  for (int i = 0; i < ngs; ++i) B.add(gs[i].p, gs[i].w, batch);
  B.add(aggR, me.dout, (const int*)0);
  B.add(aggS, me.dout, (const int*)0);
  launch_gemm(B.d, mn.din, mn.W1t, mn.b1, Hn, GN_NODES, mn.lat, Hn_ld, true, st);
  SegBuild B2; B2.add(Hn, Hn_ld, (const int*)0);
  launch_gemm(B2.d, Hn_ld, mn.W2t, mn.b2, n_out, GN_NODES, mn.dout, mn.dout,
              act_final, st);
  if (mn.ln)
    gn_layernorm128<<<(GN_NODES + 7) / 8, 256, 0, st>>>(n_out, mn.ln_g, mn.ln_b, GN_NODES);

  // ---- node/edge -> global aggregation ----
  launch_zero(n2g, (long long)GN_GRAPHS * mn.dout, st);
  launch_zero(e2g, (long long)GN_GRAPHS * me.dout, st);
  launch_scatter(n_out, batch, n2g, GN_NODES, mn.dout, st);
  launch_scatter(e_out, eb, e2g, GN_EDGES, me.dout, st);

  // ---- global MLP: concat(n2g, e2g, g) ; never LayerNorm ----
  SegBuild C;
  C.add(n2g, mn.dout, (const int*)0);
  C.add(e2g, me.dout, (const int*)0);
  for (int i = 0; i < ngs; ++i) C.add(gs[i].p, gs[i].w, (const int*)0);
  launch_gemm(C.d, mg.din, mg.W1t, mg.b1, Hg, GN_GRAPHS, mg.lat, Hg_ld, true, st);
  SegBuild C2; C2.add(Hg, Hg_ld, (const int*)0);
  launch_gemm(C2.d, Hg_ld, mg.W2t, mg.b2, g_out, GN_GRAPHS, mg.dout, mg.dout,
              act_final, st);
}

// MLP dims, in jax pytree (alphabetical) flatten order:
// decoder{edge,global,node}, encoder{...}, output{...}, processor{...}
struct MLPDims { int din, lat, dout; bool ln; };
static const MLPDims kDims[12] = {
  {512, 128, 128, true}, {384, 128, 128, false}, {512, 128, 128, true},   // decoder
  {13,  128, 128, true}, {258, 128, 128, false}, {262, 128, 128, true},   // encoder
  {512, 65,  3,  false}, {134, 65,  1,  false},  {262, 65,  3,  false},   // output
  {1024,128, 128, true}, {512, 128, 128, false}, {768, 128, 128, true},   // processor
};

extern "C" void kernel_launch(void* const* d_in, const int* in_sizes, int n_in,
                              void* d_out, int out_size, void* d_ws, size_t ws_size,
                              hipStream_t stream)
{
  (void)in_sizes; (void)n_in; (void)out_size; (void)ws_size;

  const float* edge_attr   = (const float*)d_in[0];
  const float* node_attr   = (const float*)d_in[1];
  const float* global_attr = (const float*)d_in[2];
  const int*   eidx        = (const int*)d_in[3];
  const int*   batch       = (const int*)d_in[4];
  const int*   row = eidx;
  const int*   col = eidx + GN_EDGES;

  // ---- parse params (pytree leaves: W1, W2, b1, b2, [ln_b, ln_g]) ----
  MLPW m[12];
  int ii = 5;
  for (int q = 0; q < 12; ++q) {
    m[q].din = kDims[q].din; m[q].lat = kDims[q].lat;
    m[q].dout = kDims[q].dout; m[q].ln = kDims[q].ln;
    m[q].W1 = (const float*)d_in[ii++];
    m[q].W2 = (const float*)d_in[ii++];
    m[q].b1 = (const float*)d_in[ii++];
    m[q].b2 = (const float*)d_in[ii++];
    m[q].ln_b = 0; m[q].ln_g = 0;
    if (m[q].ln) { m[q].ln_b = (const float*)d_in[ii++]; m[q].ln_g = (const float*)d_in[ii++]; }
  }

  // ---- workspace carve-up ----
  float* F = (float*)d_ws;
  const size_t EW = (size_t)GN_EDGES * 128, NW = (size_t)GN_NODES * 128;
  float* e0 = F; F += EW;  float* eA = F; F += EW;
  float* eB = F; F += EW;  float* He = F; F += EW;
  float* n0 = F; F += NW;  float* nA = F; F += NW;
  float* nB = F; F += NW;  float* Hn = F; F += NW;
  float* aggS = F; F += NW; float* aggR = F; F += NW;
  float* g0 = F; F += 1024; float* gA = F; F += 1024;
  float* gB = F; F += 1024; float* Hg = F; F += 1024;
  float* n2g = F; F += 1024; float* e2g = F; F += 1024;
  int* ebArr = (int*)F; F += GN_EDGES;
  unsigned short* wreg = (unsigned short*)F;

  // ---- convert + transpose all weights to padded bf16 [Npad][Kpad] ----
  size_t wpos = 0;
  for (int q = 0; q < 12; ++q) {
    {
      const int K = m[q].din, N = m[q].lat, Kp = rnd32(K), Np = rnd16(N);
      m[q].W1t = wreg + wpos; wpos = (wpos + (size_t)Kp * Np + 7) & ~(size_t)7;
      const long long tot = (long long)Kp * Np;
      gn_wt_cast<<<(unsigned)((tot + 255) / 256), 256, 0, stream>>>(m[q].W1, m[q].W1t, K, N, Kp, Np);
    }
    {
      const int K = m[q].lat, N = m[q].dout, Kp = rnd32(K), Np = rnd16(N);
      m[q].W2t = wreg + wpos; wpos = (wpos + (size_t)Kp * Np + 7) & ~(size_t)7;
      const long long tot = (long long)Kp * Np;
      gn_wt_cast<<<(unsigned)((tot + 255) / 256), 256, 0, stream>>>(m[q].W2, m[q].W2t, K, N, Kp, Np);
    }
  }

  gn_edge_batch<<<(GN_EDGES + 255) / 256, 256, 0, stream>>>(batch, row, ebArr, GN_EDGES);

  // ---- encoder: raw attrs -> (e0, n0, g0) ----
  {
    Seg ns[1] = {{node_attr, 4}};
    Seg es[1] = {{edge_attr, 3}};
    Seg gs[1] = {{global_attr, 2}};
    run_block(m[3], m[5], m[4], ns, 1, es, 1, gs, 1, row, col, batch, ebArr,
              e0, n0, g0, He, Hn, Hg, aggS, aggR, n2g, e2g, true, stream);
  }

  // ---- processor x3 on concat([e0,e],[n0,n],[g0,g]) ----
  float* ebufs[2] = {eA, eB};
  float* nbufs[2] = {nA, nB};
  float* gbufs[2] = {gA, gB};
  float *ce = e0, *cn = n0, *cg = g0;
  for (int t = 0; t < 3; ++t) {
    float* oe = ebufs[t & 1];
    float* on = nbufs[t & 1];
    float* og = gbufs[t & 1];
    Seg ns[2] = {{n0, 128}, {cn, 128}};
    Seg es[2] = {{e0, 128}, {ce, 128}};
    Seg gs[2] = {{g0, 128}, {cg, 128}};
    run_block(m[9], m[11], m[10], ns, 2, es, 2, gs, 2, row, col, batch, ebArr,
              oe, on, og, He, Hn, Hg, aggS, aggR, n2g, e2g, true, stream);
    ce = oe; cn = on; cg = og;
  }

  // Only the final iteration's decoder/output affect the result.
  float* de = ebufs[1];   // free buffer after 3 swaps (last write went to [0])
  float* dn = nbufs[1];
  float* dg = gbufs[1];
  {
    Seg ns[1] = {{cn, 128}};
    Seg es[1] = {{ce, 128}};
    Seg gs[1] = {{cg, 128}};
    run_block(m[0], m[2], m[1], ns, 1, es, 1, gs, 1, row, col, batch, ebArr,
              de, dn, dg, He, Hn, Hg, aggS, aggR, n2g, e2g, true, stream);
  }

  // ---- output block writes straight into d_out: e(64000x3) | n(16000x3) | g(8x1)
  {
    float* oe = (float*)d_out;
    float* on = oe + (size_t)GN_EDGES * 3;
    float* og = on + (size_t)GN_NODES * 3;
    Seg ns[1] = {{dn, 128}};
    Seg es[1] = {{de, 128}};
    Seg gs[1] = {{dg, 128}};
    run_block(m[6], m[8], m[7], ns, 1, es, 1, gs, 1, row, col, batch, ebArr,
              oe, on, og, He, Hn, Hg, aggS, aggR, n2g, e2g, false, stream);
  }
}
